// FWMRNN_25675314495567
// MI455X (gfx1250) — matrix operand order repaired and checked by reading, not compiled, not run on hardware
//
#include <hip/hip_runtime.h>
#include <stddef.h>

// ---------------- problem constants ----------------
#define T_STEPS 2048
#define BATCH   32
#define ISIZE   256
#define HSIZE   512
#define KSIZE   64
#define VSIZE   64
#define MROWS   (T_STEPS * BATCH)      // 65536
#define NWG_LSTM 32
#define HSL      16                    // H columns per LSTM workgroup

#define AS1 __attribute__((address_space(1)))
#define AS3 __attribute__((address_space(3)))

#if __has_builtin(__builtin_amdgcn_global_load_async_to_lds_b128) && \
    __has_builtin(__builtin_amdgcn_s_wait_asynccnt)
#define HAVE_ASYNC_LDS 1
#else
#define HAVE_ASYNC_LDS 0
#endif

// ---------------- WMMA types ----------------
typedef __attribute__((ext_vector_type(16))) __bf16 bf16x16;
typedef __attribute__((ext_vector_type(8)))  float  f32x8;
typedef __attribute__((ext_vector_type(4)))  unsigned u32x4;
typedef int i32x4 __attribute__((vector_size(16)));   // matches async-LDS builtin pointee
typedef unsigned short u16;

struct bf16frag { u32x4 lo, hi; };

static __device__ __forceinline__ u16 f2bf(float f) {
    unsigned u = __builtin_bit_cast(unsigned, f);
    unsigned r = u + 0x7FFFu + ((u >> 16) & 1u);   // round-to-nearest-even
    return (u16)(r >> 16);
}

static __device__ __forceinline__ float sigmoidf(float x) {
    return 1.0f / (1.0f + __expf(-x));
}

// A-matrix fragment (16x32 bf16, row-major source [rows, ld]).
// ISA layout: lanes 0-15 -> row M=lane, K = k0+0..7 and k0+16..23;
//             lanes 16-31 -> row M=lane-16, K = k0+8..15 and k0+24..31.
static __device__ __forceinline__ bf16x16
load_a_frag(const u16* p, int ld, int row0, int k0) {
    int lane = threadIdx.x & 31;
    int half = lane >> 4;
    const u16* q = p + (size_t)(row0 + (lane & 15)) * ld + k0 + half * 8;
    bf16frag f;
    f.lo = *(const u32x4*)(q);
    f.hi = *(const u32x4*)(q + 16);
    return __builtin_bit_cast(bf16x16, f);
}

// B-matrix fragment (32x16 bf16). B[k][n] = W[n][k] with W row-major [N, Kdim].
// Each lane reads K = k0+8h..+7 and k0+16+8h..+7 from W's row n -> two b128 loads.
static __device__ __forceinline__ bf16x16
load_b_frag(const u16* __restrict__ p, int ld, int n0, int k0) {
    int lane = threadIdx.x & 31;
    const u16* q = p + (size_t)(n0 + (lane & 15)) * ld + k0 + (lane >> 4) * 8;
    bf16frag f;
    f.lo = *(const u32x4*)(q);
    f.hi = *(const u32x4*)(q + 16);
    return __builtin_bit_cast(bf16x16, f);
}

static __device__ __forceinline__ f32x8
wmma_bf16(bf16x16 a, bf16x16 b, f32x8 c) {
    return __builtin_amdgcn_wmma_f32_16x16x32_bf16(
        false, a, false, b, (short)0, c, false, false);
}

// ---------------- fp32 -> bf16 conversion ----------------
__global__ void k_f2bf(const float* __restrict__ in, u16* __restrict__ o, int n) {
    int i = blockIdx.x * blockDim.x + threadIdx.x;
    if (i < n) o[i] = f2bf(in[i]);
}

// ---------------- init: h ping-pong buffer + sync counter ----------------
__global__ void k_init(const float* __restrict__ h0, u16* __restrict__ hbuf,
                       int* __restrict__ cnt) {
    int i = blockIdx.x * blockDim.x + threadIdx.x;
    if (i < BATCH * HSIZE) hbuf[i] = f2bf(h0[i]);
    if (i == 0) *cnt = 0;
}

// ---------------- persistent LSTM over time ----------------
// 32 WGs; WG wg owns H-slice [wg*16, wg*16+16). 8 waves: wave = (g<<1)|m,
// m = batch half, g = gate (i,f,g,o). Weight B-fragments stay in registers
// for all 2048 steps; the x tile is double-buffered into LDS with async
// loads issued one step ahead; h ping-pongs through a global bf16 buffer
// guarded by an agent-scope atomic grid barrier.
__global__ __launch_bounds__(256) void k_lstm(
    const u16*  __restrict__ x_bf,     // [T*B, ISIZE] bf16
    const u16*  __restrict__ wih_bf,   // [4H, ISIZE]  bf16
    const u16*  __restrict__ whh_bf,   // [4H, HSIZE]  bf16
    const float* __restrict__ b_ih,    // [4H]
    const float* __restrict__ b_hh,    // [4H]
    const float* __restrict__ c0,      // [B, H]
    u16*   __restrict__ hbuf,          // [2][B*H] bf16 ping-pong
    u16*   __restrict__ hseq_bf,       // [T*B, H] bf16
    float* __restrict__ out,           // [T*B, 2H] fp32 (first half)
    int*   __restrict__ sync_cnt)
{
    __shared__ float gates_s[4][BATCH][HSL];
    __shared__ float c_s[BATCH][HSL];
    __shared__ u16   xbuf[2][BATCH][ISIZE];    // 2 x 16KB

    const int wg   = blockIdx.x;
    const int h0c  = wg * HSL;
    const int tid  = threadIdx.x;
    const int wave = tid >> 5;
    const int lane = tid & 31;
    const int m    = wave & 1;
    const int g    = wave >> 1;

    // init cell state slice
    for (int i = tid; i < BATCH * HSL; i += 256) {
        int b = i >> 4, j = i & 15;
        c_s[b][j] = c0[b * HSIZE + h0c + j];
    }

    // preload weight B-fragments into registers (persist across all steps)
    const int n0 = g * HSIZE + h0c;
    bf16x16 wih_f[ISIZE / 32];
#pragma unroll
    for (int kk = 0; kk < ISIZE / 32; ++kk)
        wih_f[kk] = load_b_frag(wih_bf, ISIZE, n0, kk * 32);
    bf16x16 whh_f[HSIZE / 32];
#pragma unroll
    for (int kk = 0; kk < HSIZE / 32; ++kk)
        whh_f[kk] = load_b_frag(whh_bf, HSIZE, n0, kk * 32);

    const float bcol = b_ih[n0 + (lane & 15)] + b_hh[n0 + (lane & 15)];

    // stage x tile for step t into LDS buffer `buf` (16KB: 4 b128 per thread)
    auto issue_x = [&](int t, int buf) {
        const char* src = (const char*)(x_bf + (size_t)t * BATCH * ISIZE);
        char* dst = (char*)&xbuf[buf][0][0];
#pragma unroll
        for (int i = 0; i < 4; ++i) {
            int byteoff = (tid + i * 256) * 16;
#if HAVE_ASYNC_LDS
            __builtin_amdgcn_global_load_async_to_lds_b128(
                (AS1 i32x4*)(src + byteoff), (AS3 i32x4*)(dst + byteoff), 0, 0);
#else
            *(u32x4*)(dst + byteoff) = *(const u32x4*)(src + byteoff);
#endif
        }
    };

    issue_x(0, 0);
#if HAVE_ASYNC_LDS
    __builtin_amdgcn_s_wait_asynccnt(0);
#endif
    __syncthreads();

    for (int t = 0; t < T_STEPS; ++t) {
        // async-stage next step's x tile while this step computes
        if (t + 1 < T_STEPS) issue_x(t + 1, (t + 1) & 1);
        // pull the tile after next into L2 ahead of the async engine
        if (t + 2 < T_STEPS)
            __builtin_prefetch(x_bf + (size_t)(t + 2) * BATCH * ISIZE + tid * 32,
                               0, 1);

        const u16* hb   = hbuf + (size_t)(t & 1) * (BATCH * HSIZE);
        const u16* xlds = (const u16*)&xbuf[t & 1][0][0];

        // two independent accumulation chains (x-path and h-path)
        f32x8 accx = {}, acch = {};
#pragma unroll
        for (int kk = 0; kk < ISIZE / 32; ++kk) {
            bf16x16 a = load_a_frag(xlds, ISIZE, m * 16, kk * 32);
            accx = wmma_bf16(a, wih_f[kk], accx);
        }
#pragma unroll
        for (int kk = 0; kk < HSIZE / 32; ++kk) {
            bf16x16 a = load_a_frag(hb, HSIZE, m * 16, kk * 32);
            acch = wmma_bf16(a, whh_f[kk], acch);
        }

        // scatter 16x16 gate tile into LDS (C layout: N=lane&15, M=(lane>>4)*8+r)
        {
            int N = lane & 15, Mb = (lane >> 4) * 8;
#pragma unroll
            for (int r = 0; r < 8; ++r)
                gates_s[g][m * 16 + Mb + r][N] = accx[r] + acch[r] + bcol;
        }
#if HAVE_ASYNC_LDS
        __builtin_amdgcn_s_wait_asynccnt(0);
#endif
        __syncthreads();

        // elementwise LSTM cell: 32x16 cells, 2 per thread
#pragma unroll
        for (int e = 0; e < 2; ++e) {
            int cell = tid * 2 + e;
            int b = cell >> 4, j = cell & 15;
            float ig = sigmoidf(gates_s[0][b][j]);
            float fg = sigmoidf(gates_s[1][b][j]);
            float gg = tanhf(gates_s[2][b][j]);
            float og = sigmoidf(gates_s[3][b][j]);
            float c  = fg * c_s[b][j] + ig * gg;
            float h  = og * tanhf(c);
            c_s[b][j] = c;
            size_t row = (size_t)t * BATCH + b;
            out[row * (2 * HSIZE) + h0c + j] = h;
            u16 hb16 = f2bf(h);
            hseq_bf[row * HSIZE + h0c + j] = hb16;
            hbuf[(size_t)((t + 1) & 1) * (BATCH * HSIZE) + b * HSIZE + h0c + j] = hb16;
        }
        __threadfence();
        __syncthreads();

        // grid-wide barrier: monotonic arrive counter
        if (tid == 0) {
            __hip_atomic_fetch_add(sync_cnt, 1, __ATOMIC_ACQ_REL,
                                   __HIP_MEMORY_SCOPE_AGENT);
            int target = NWG_LSTM * (t + 1);
            while (__hip_atomic_load(sync_cnt, __ATOMIC_ACQUIRE,
                                     __HIP_MEMORY_SCOPE_AGENT) < target) {
                __builtin_amdgcn_s_sleep(8);
            }
        }
        __syncthreads();
    }
}

// ---------------- fused k/v/q projections: one A pass, three WMMA chains ----------
// [M,512] x ([512,64] x3) -> ks,vs,qs [M,64] (+bias)
__global__ __launch_bounds__(256) void k_gemm_kvq(
    const u16*  __restrict__ A,       // hseq bf16 [M, 512]
    const u16*  __restrict__ wk16, const u16* __restrict__ wv16,
    const u16*  __restrict__ wq16,
    const float* __restrict__ bk, const float* __restrict__ bv,
    const float* __restrict__ bq,
    float* __restrict__ ko, float* __restrict__ vo, float* __restrict__ qo)
{
    int wave = threadIdx.x >> 5, lane = threadIdx.x & 31;
    int m = wave & 1, nt = wave >> 1;
    int row0 = blockIdx.x * 32 + m * 16;
    int n0 = nt * 16;
    f32x8 ak = {}, av = {}, aq = {};
#pragma unroll
    for (int kk = 0; kk < HSIZE / 32; ++kk) {
        bf16x16 a = load_a_frag(A, HSIZE, row0, kk * 32);
        ak = wmma_bf16(a, load_b_frag(wk16, HSIZE, n0, kk * 32), ak);
        av = wmma_bf16(a, load_b_frag(wv16, HSIZE, n0, kk * 32), av);
        aq = wmma_bf16(a, load_b_frag(wq16, HSIZE, n0, kk * 32), aq);
    }
    int N = lane & 15, Mb = (lane >> 4) * 8;
    float bkv = bk[n0 + N], bvv = bv[n0 + N], bqv = bq[n0 + N];
#pragma unroll
    for (int r = 0; r < 8; ++r) {
        size_t o = (size_t)(row0 + Mb + r) * 64 + n0 + N;
        ko[o] = ak[r] + bkv;
        vo[o] = av[r] + bvv;
        qo[o] = aq[r] + bqv;
    }
}

// ---------------- elu(x)+1 then row-normalize over 64 cols (in place) -------------
__global__ __launch_bounds__(256) void k_elu_norm(float* __restrict__ x)
{
    int wave = threadIdx.x >> 5, lane = threadIdx.x & 31;
    int row = blockIdx.x * 8 + wave;
    float* p = x + (size_t)row * 64;
    float a = p[lane], b = p[lane + 32];
    a = a > 0.f ? a + 1.f : __expf(a);   // elu(x)+1
    b = b > 0.f ? b + 1.f : __expf(b);
    float s = a + b;
#pragma unroll
    for (int o = 16; o >= 1; o >>= 1) s += __shfl_xor(s, o, 32);
    float inv = 1.f / s;
    p[lane] = a * inv;
    p[lane + 32] = b * inv;
}

// ---------------- scalar gate projections: bs = sig(h.Wb+bb), ms = sig(h.Wm+bm) ---
__global__ __launch_bounds__(256) void k_bm(
    const float* __restrict__ out_h,  // d_out, [M, 1024], first 512 cols = h
    const float* __restrict__ Wb, const float* __restrict__ bb,
    const float* __restrict__ Wm, const float* __restrict__ bm,
    float* __restrict__ bs, float* __restrict__ ms)
{
    int wave = threadIdx.x >> 5, lane = threadIdx.x & 31;
    int row = blockIdx.x * 8 + wave;
    const float* h = out_h + (size_t)row * (2 * HSIZE);
    float sb = 0.f, sm = 0.f;
#pragma unroll
    for (int i = 0; i < 16; ++i) {
        float hv = h[lane * 16 + i];
        sb = fmaf(hv, Wb[lane * 16 + i], sb);
        sm = fmaf(hv, Wm[lane * 16 + i], sm);
    }
#pragma unroll
    for (int o = 16; o >= 1; o >>= 1) {
        sb += __shfl_xor(sb, o, 32);
        sm += __shfl_xor(sm, o, 32);
    }
    if (lane == 0) {
        bs[row] = sigmoidf(sb + bb[0]);
        ms[row] = sigmoidf(sm + bm[0]);
    }
}

// ---------------- delta-rule fast-weight scan: one WG per batch -------------------
__global__ __launch_bounds__(64) void k_fwm(
    const float* __restrict__ F0,   // [B, 64, 64]
    const float* __restrict__ ks,   // [M, 64] (normalized)
    const float* __restrict__ vs,   // [M, 64]
    const float* __restrict__ qs,   // [M, 64] (normalized)
    const float* __restrict__ bs,   // [M]
    const float* __restrict__ ms,   // [M]
    u16* __restrict__ s_bf)         // [M, 64] bf16
{
    int b = blockIdx.x;
    int v = threadIdx.x;            // 0..63 -> F row
    float F[64];
#pragma unroll
    for (int j = 0; j < 64; ++j) F[j] = F0[((size_t)b * 64 + v) * 64 + j];

    __shared__ float k_s[64], mq_s[64];
    for (int t = 0; t < T_STEPS; ++t) {
        size_t row = (size_t)t * BATCH + b;
        float mt = ms[row];
        k_s[v]  = ks[row * 64 + v];
        mq_s[v] = mt * qs[row * 64 + v];
        __syncthreads();
        float vold = 0.f;
#pragma unroll
        for (int j = 0; j < 64; ++j) vold = fmaf(F[j], k_s[j], vold);
        float nv = bs[row] * (vs[row * 64 + v] - vold);
        float o = 0.f;
#pragma unroll
        for (int j = 0; j < 64; ++j) {
            F[j] = fmaf(nv, k_s[j], F[j]);
            o = fmaf(F[j], mq_s[j], o);
        }
        s_bf[row * 64 + v] = f2bf(o);
        __syncthreads();
    }
}

// ---------------- output projection: [M,64]x[64,512] -> out[:,512:1024] -----------
__global__ __launch_bounds__(256) void k_gemm_out(
    const u16*  __restrict__ S,      // [M, 64] bf16
    const u16*  __restrict__ Wl,     // [512, 64] bf16 row-major
    const float* __restrict__ blin,  // [512]
    float* __restrict__ out)         // [M, 1024]
{
    int wave = threadIdx.x >> 5, lane = threadIdx.x & 31;
    int m = wave & 1, nt = wave >> 1;
    int row0 = blockIdx.x * 32 + m * 16;
    int n0 = blockIdx.y * 64 + nt * 16;
    f32x8 acc = {};
#pragma unroll
    for (int kk = 0; kk < 2; ++kk) {
        bf16x16 a = load_a_frag(S, 64, row0, kk * 32);
        bf16x16 b = load_b_frag(Wl, 64, n0, kk * 32);
        acc = wmma_bf16(a, b, acc);
    }
    int N = lane & 15, Mb = (lane >> 4) * 8;
    float bv = blin[n0 + N];
#pragma unroll
    for (int r = 0; r < 8; ++r)
        out[(size_t)(row0 + Mb + r) * (2 * HSIZE) + HSIZE + n0 + N] = acc[r] + bv;
}

// ---------------- host orchestration ----------------
extern "C" void kernel_launch(void* const* d_in, const int* in_sizes, int n_in,
                              void* d_out, int out_size, void* d_ws, size_t ws_size,
                              hipStream_t stream) {
    (void)in_sizes; (void)n_in; (void)out_size; (void)ws_size;
    const float* inputs = (const float*)d_in[0];
    const float* h0     = (const float*)d_in[1];
    const float* c0     = (const float*)d_in[2];
    const float* F0     = (const float*)d_in[3];
    const float* W_ih   = (const float*)d_in[4];
    const float* W_hh   = (const float*)d_in[5];
    const float* b_ih   = (const float*)d_in[6];
    const float* b_hh   = (const float*)d_in[7];
    const float* Wk     = (const float*)d_in[8];
    const float* bk     = (const float*)d_in[9];
    const float* Wv     = (const float*)d_in[10];
    const float* bv     = (const float*)d_in[11];
    const float* Wb     = (const float*)d_in[12];
    const float* bb     = (const float*)d_in[13];
    const float* Wq     = (const float*)d_in[14];
    const float* bq     = (const float*)d_in[15];
    const float* Wm     = (const float*)d_in[16];
    const float* bm     = (const float*)d_in[17];
    const float* Wlin   = (const float*)d_in[18];
    const float* blin   = (const float*)d_in[19];
    float* out = (float*)d_out;

    char* ws = (char*)d_ws;
    size_t off = 0;
    auto alloc = [&](size_t bytes) -> void* {
        void* p = ws + off;
        off = (off + bytes + 255) & ~(size_t)255;
        return p;
    };

    u16*   x_bf    = (u16*)  alloc((size_t)MROWS * ISIZE * 2);
    u16*   wih_bf  = (u16*)  alloc((size_t)4 * HSIZE * ISIZE * 2);
    u16*   whh_bf  = (u16*)  alloc((size_t)4 * HSIZE * HSIZE * 2);
    u16*   wk_bf   = (u16*)  alloc((size_t)KSIZE * HSIZE * 2);
    u16*   wv_bf   = (u16*)  alloc((size_t)VSIZE * HSIZE * 2);
    u16*   wq_bf   = (u16*)  alloc((size_t)KSIZE * HSIZE * 2);
    u16*   wlin_bf = (u16*)  alloc((size_t)HSIZE * VSIZE * 2);
    u16*   hbuf    = (u16*)  alloc((size_t)2 * BATCH * HSIZE * 2);
    u16*   hseq_bf = (u16*)  alloc((size_t)MROWS * HSIZE * 2);
    float* ks_ws   = (float*)alloc((size_t)MROWS * KSIZE * 4);
    float* vs_ws   = (float*)alloc((size_t)MROWS * VSIZE * 4);
    float* qs_ws   = (float*)alloc((size_t)MROWS * KSIZE * 4);
    float* bs_ws   = (float*)alloc((size_t)MROWS * 4);
    float* ms_ws   = (float*)alloc((size_t)MROWS * 4);
    u16*   s_bf    = (u16*)  alloc((size_t)MROWS * VSIZE * 2);
    int*   sync_c  = (int*)  alloc(256);

    // 1. bf16 conversions
    {
        int n = MROWS * ISIZE;
        k_f2bf<<<(n + 255) / 256, 256, 0, stream>>>(inputs, x_bf, n);
        n = 4 * HSIZE * ISIZE;
        k_f2bf<<<(n + 255) / 256, 256, 0, stream>>>(W_ih, wih_bf, n);
        n = 4 * HSIZE * HSIZE;
        k_f2bf<<<(n + 255) / 256, 256, 0, stream>>>(W_hh, whh_bf, n);
        n = KSIZE * HSIZE;
        k_f2bf<<<(n + 255) / 256, 256, 0, stream>>>(Wk, wk_bf, n);
        k_f2bf<<<(n + 255) / 256, 256, 0, stream>>>(Wv, wv_bf, n);
        k_f2bf<<<(n + 255) / 256, 256, 0, stream>>>(Wq, wq_bf, n);
        n = HSIZE * VSIZE;
        k_f2bf<<<(n + 255) / 256, 256, 0, stream>>>(Wlin, wlin_bf, n);
    }

    // 2. init h ping-pong + barrier counter
    k_init<<<(BATCH * HSIZE + 255) / 256, 256, 0, stream>>>(h0, hbuf, sync_c);

    // 3. persistent LSTM (writes h into out[:, :512] and hseq_bf)
    k_lstm<<<NWG_LSTM, 256, 0, stream>>>(x_bf, wih_bf, whh_bf, b_ih, b_hh, c0,
                                         hbuf, hseq_bf, out, sync_c);

    // 4. fused FWM projections + activations
    k_gemm_kvq<<<MROWS / 32, 256, 0, stream>>>(hseq_bf, wk_bf, wv_bf, wq_bf,
                                               bk, bv, bq, ks_ws, vs_ws, qs_ws);
    k_elu_norm<<<MROWS / 8, 256, 0, stream>>>(ks_ws);
    k_elu_norm<<<MROWS / 8, 256, 0, stream>>>(qs_ws);
    k_bm<<<MROWS / 8, 256, 0, stream>>>(out, Wb, bb, Wm, bm, bs_ws, ms_ws);

    // 5. delta-rule fast-weight scan (one WG per batch)
    k_fwm<<<BATCH, 64, 0, stream>>>(F0, ks_ws, vs_ws, qs_ws, bs_ws, ms_ws, s_bf);

    // 6. output projection into out[:, 512:1024]
    k_gemm_out<<<dim3(MROWS / 32, HSIZE / 64), 256, 0, stream>>>(s_bf, wlin_bf,
                                                                 blin, out);
}
